// EdgeMLP_51170240365037
// MI455X (gfx1250) — compile-verified, matrix operand order checked by reading, not run on hardware
//
#include <hip/hip_runtime.h>
#include <hip/hip_bf16.h>

typedef __attribute__((ext_vector_type(16))) _Float16 v16h;
typedef __attribute__((ext_vector_type(8)))  float    v8f;

#define HID      128
#define N_EDGES  400000
#define N_TILES  25000   // N_EDGES / 16
#define NBLK     2500    // blocks in main kernel; 10 tiles each

// ws layout (bytes):
//   [0        .. 131072)  W11 packed f16, WMMA-B layout (65536 halves)
//   [131072   .. 163840)  W12 packed f16 (16384 halves)
//   [163840   .. 196608)  W13 packed f16 (16384 halves)
//   [196608   .. 2756608) per-block partial sums: NBLK x {sum[128], sumsq[128]}
//   [2756608  .. 2757632) scale[128], shift[128]
#define PART_OFF 196608
#define SS_OFF   2756608

__device__ __forceinline__ float gelu_exact(float x) {
    return 0.5f * x * (1.0f + erff(x * 0.70710678118654752f));
}

// ---------------------------------------------------------------------------
// Kernel 0: pack f32 weights -> f16 in WMMA B-operand layout.
// Linear index: ((ks*8 + nt)*32 + lane)*16 + h
//   k = ks*32 + ((h&8)?16:0) + ((lane>=16)?8:0) + (h&7)
//   n = nt*16 + (lane&15)
// ---------------------------------------------------------------------------
__global__ void edge_mlp_pack(const float* __restrict__ W11,
                              const float* __restrict__ W12,
                              const float* __restrict__ W13,
                              _Float16* __restrict__ wp) {
    const int g = blockIdx.x * 256 + threadIdx.x;   // 98304 total
    const float* W;
    int idx;
    if (g < 65536)      { W = W11; idx = g; }
    else if (g < 81920) { W = W12; idx = g - 65536; }
    else                { W = W13; idx = g - 81920; }
    const int h    = idx & 15;
    const int ln   = (idx >> 4) & 31;
    const int ntks = idx >> 9;
    const int nt   = ntks & 7;
    const int ks   = ntks >> 3;
    const int k  = (ks << 5) + ((h & 8) ? 16 : 0) + ((ln >> 4) << 3) + (h & 7);
    const int nn = (nt << 4) + (ln & 15);
    wp[g] = (_Float16)W[k * HID + nn];
}

// ---------------------------------------------------------------------------
// Kernel 1: fused gather + 3-layer MLP (WMMA f16) + residual; writes x to out
// and per-block feature partial sums to ws.
// ---------------------------------------------------------------------------
__global__ __launch_bounds__(256) void edge_mlp_main(
    const float* __restrict__ h_V, const float* __restrict__ h_E,
    const int*   __restrict__ edge_idx, const float* __restrict__ vff,
    const _Float16* __restrict__ wp,
    const float* __restrict__ b11, const float* __restrict__ b12,
    const float* __restrict__ b13,
    float* __restrict__ out, float* __restrict__ part)
{
    __shared__ __align__(32) _Float16 As1[16 * 512];  // 16 KB, layer-1 A tiles
    __shared__ __align__(32) _Float16 As2[16 * 128];  // 4 KB
    __shared__ __align__(32) _Float16 As3[16 * 128];  // 4 KB
    __shared__ float sRed[256];                       // sum[128], sumsq[128]

    const int tid  = threadIdx.x;
    const int lane = tid & 31;
    const int wave = tid >> 5;                        // N-slab 0..7

    const _Float16* w11p = wp;
    const _Float16* w12p = wp + 65536;
    const _Float16* w13p = wp + 81920;
    const int* srcI = edge_idx;
    const int* dstI = edge_idx + N_EDGES;

    sRed[tid] = 0.0f;
    __syncthreads();

    const int n  = (wave << 4) | (lane & 15);         // output feature column
    const int mb = (lane >> 4) << 3;                  // row offset of D rows
    // LDS A-layout slot for writing activation element (m, k=n):
    const int wlane = ((n >> 3) & 1) << 4;            // +16 if k bit3
    const int wh    = (((n >> 4) & 1) << 3) | (n & 7);
    const int wks   = n >> 5;
    const float bias1 = b11[n];
    const float bias2 = b12[n];
    const float bias3 = b13[n];

    float lSum = 0.0f, lSq = 0.0f;

    for (int t = blockIdx.x; t < N_TILES; t += NBLK) {
        const int e0 = t << 4;

        // ---- stage layer-1 A: gather/concat 16x512 f32 -> f16 A-layout ----
        for (int i = tid; i < 512; i += 256) {        // i = ks*32 + lane
            const int ks = i >> 5;
            const int ln = i & 31;
            const int m  = ln & 15;
            const int kb = (ln >> 4) << 3;            // 0 or 8
            const int e  = e0 + m;
            const int seg = ks >> 2;                  // 128-wide concat segment
            const float* row;
            if (seg == 0)      row = h_V + (size_t)srcI[e] * HID;
            else if (seg == 1) row = h_E + (size_t)e * HID;
            else if (seg == 2) row = h_V + (size_t)dstI[e] * HID;
            else               row = vff + (size_t)e * HID;
            const float* base = row + ((ks & 3) << 5) + kb;
            const float4 f0 = ((const float4*)base)[0];
            const float4 f1 = ((const float4*)base)[1];
            const float4 f2 = ((const float4*)(base + 16))[0];
            const float4 f3 = ((const float4*)(base + 16))[1];
            _Float16* o = &As1[i << 4];
            o[0]=(_Float16)f0.x;  o[1]=(_Float16)f0.y;  o[2]=(_Float16)f0.z;  o[3]=(_Float16)f0.w;
            o[4]=(_Float16)f1.x;  o[5]=(_Float16)f1.y;  o[6]=(_Float16)f1.z;  o[7]=(_Float16)f1.w;
            o[8]=(_Float16)f2.x;  o[9]=(_Float16)f2.y;  o[10]=(_Float16)f2.z; o[11]=(_Float16)f2.w;
            o[12]=(_Float16)f3.x; o[13]=(_Float16)f3.y; o[14]=(_Float16)f3.z; o[15]=(_Float16)f3.w;
        }
        __syncthreads();

        // ---- layer 1: [16x512] x [512x16] per wave, 16 K-steps ----
        v8f acc;
        #pragma unroll
        for (int r = 0; r < 8; r++) acc[r] = 0.0f;
        #pragma unroll
        for (int ks = 0; ks < 16; ks++) {
            v16h a = *(const v16h*)&As1[(((ks << 5) | lane)) << 4];
            v16h b = *(const v16h*)&w11p[((((ks << 3) | wave) << 5) | lane) << 4];
            acc = __builtin_amdgcn_wmma_f32_16x16x32_f16(false, a, false, b,
                                                         (short)0, acc, false, false);
        }
        #pragma unroll
        for (int r = 0; r < 8; r++) {
            float g = gelu_exact(acc[r] + bias1);
            As2[(((wks << 5) + mb + r + wlane) << 4) | wh] = (_Float16)g;
        }
        __syncthreads();

        // ---- layer 2: 4 K-steps ----
        v8f acc2;
        #pragma unroll
        for (int r = 0; r < 8; r++) acc2[r] = 0.0f;
        #pragma unroll
        for (int ks = 0; ks < 4; ks++) {
            v16h a = *(const v16h*)&As2[(((ks << 5) | lane)) << 4];
            v16h b = *(const v16h*)&w12p[((((ks << 3) | wave) << 5) | lane) << 4];
            acc2 = __builtin_amdgcn_wmma_f32_16x16x32_f16(false, a, false, b,
                                                          (short)0, acc2, false, false);
        }
        #pragma unroll
        for (int r = 0; r < 8; r++) {
            float g = gelu_exact(acc2[r] + bias2);
            As3[(((wks << 5) + mb + r + wlane) << 4) | wh] = (_Float16)g;
        }
        __syncthreads();

        // ---- layer 3 + residual + partial stats ----
        v8f acc3;
        #pragma unroll
        for (int r = 0; r < 8; r++) acc3[r] = 0.0f;
        #pragma unroll
        for (int ks = 0; ks < 4; ks++) {
            v16h a = *(const v16h*)&As3[(((ks << 5) | lane)) << 4];
            v16h b = *(const v16h*)&w13p[((((ks << 3) | wave) << 5) | lane) << 4];
            acc3 = __builtin_amdgcn_wmma_f32_16x16x32_f16(false, a, false, b,
                                                          (short)0, acc3, false, false);
        }
        #pragma unroll
        for (int r = 0; r < 8; r++) {
            const int e = e0 + mb + r;
            const size_t off = (size_t)e * HID + n;
            const float x = acc3[r] + bias3 + h_E[off];
            out[off] = x;
            lSum += x;
            lSq  += x * x;
        }
    }

    atomicAdd(&sRed[n], lSum);
    atomicAdd(&sRed[HID + n], lSq);
    __syncthreads();
    part[(size_t)blockIdx.x * 256 + tid] = sRed[tid];
}

// ---------------------------------------------------------------------------
// Kernel 2: reduce partials -> per-feature scale/shift
// ---------------------------------------------------------------------------
__global__ void edge_mlp_stats(const float* __restrict__ part,
                               const float* __restrict__ gamma,
                               const float* __restrict__ beta,
                               float* __restrict__ ss) {
    const int nn = threadIdx.x;
    if (nn >= HID) return;
    float s = 0.0f, q = 0.0f;
    for (int b = 0; b < NBLK; b++) {
        s += part[(size_t)b * 256 + nn];
        q += part[(size_t)b * 256 + HID + nn];
    }
    const float inv_e = 1.0f / (float)N_EDGES;
    const float mean = s * inv_e;
    const float var  = q * inv_e - mean * mean;
    const float sc   = rsqrtf(var + 1e-5f) * gamma[nn];
    ss[nn]       = sc;
    ss[HID + nn] = beta[nn] - mean * sc;
}

// ---------------------------------------------------------------------------
// Kernel 3: in-place affine normalize, float4 vectorized (exact grid)
// ---------------------------------------------------------------------------
__global__ void edge_mlp_norm(float* __restrict__ out,
                              const float* __restrict__ ss) {
    const size_t i = (size_t)blockIdx.x * 256 + threadIdx.x; // float4 index
    float4* p = (float4*)out;
    const int n0 = (int)((i << 2) & (HID - 1));
    float4 x = p[i];
    x.x = x.x * ss[n0 + 0] + ss[HID + n0 + 0];
    x.y = x.y * ss[n0 + 1] + ss[HID + n0 + 1];
    x.z = x.z * ss[n0 + 2] + ss[HID + n0 + 2];
    x.w = x.w * ss[n0 + 3] + ss[HID + n0 + 3];
    p[i] = x;
}

// ---------------------------------------------------------------------------
extern "C" void kernel_launch(void* const* d_in, const int* in_sizes, int n_in,
                              void* d_out, int out_size, void* d_ws, size_t ws_size,
                              hipStream_t stream) {
    (void)in_sizes; (void)n_in; (void)out_size; (void)ws_size;
    const float* h_V      = (const float*)d_in[0];
    const float* h_E      = (const float*)d_in[1];
    const int*   edge_idx = (const int*)  d_in[2];
    // d_in[3] = batch_id (unused; single batch)
    const float* vff      = (const float*)d_in[4];
    const float* W11      = (const float*)d_in[5];
    const float* b11      = (const float*)d_in[6];
    const float* W12      = (const float*)d_in[7];
    const float* b12      = (const float*)d_in[8];
    const float* W13      = (const float*)d_in[9];
    const float* b13      = (const float*)d_in[10];
    const float* gamma    = (const float*)d_in[11];
    const float* beta     = (const float*)d_in[12];
    float* out = (float*)d_out;

    char* ws = (char*)d_ws;
    _Float16* wp   = (_Float16*)ws;
    float*    part = (float*)(ws + PART_OFF);
    float*    ss   = (float*)(ws + SS_OFF);

    edge_mlp_pack <<<384,   256, 0, stream>>>(W11, W12, W13, wp);
    edge_mlp_main <<<NBLK,  256, 0, stream>>>(h_V, h_E, edge_idx, vff, wp,
                                              b11, b12, b13, out, part);
    edge_mlp_stats<<<1,     128, 0, stream>>>(part, gamma, beta, ss);
    edge_mlp_norm <<<50000, 256, 0, stream>>>(out, ss);
}
